// GeodesicSolver_111669149951
// MI455X (gfx1250) — compile-verified, matrix operand order checked by reading, not run on hardware
//
#include <hip/hip_runtime.h>

typedef __bf16 bf16;
typedef __attribute__((ext_vector_type(8)))  __bf16 v8bf;
typedef __attribute__((ext_vector_type(16))) __bf16 v16bf;
typedef __attribute__((ext_vector_type(8)))  float  v8f;

#define B_ROWS   1000
#define B_PAD    1008
#define MDIM     32
#define H1DIM    512
#define H2DIM    2048
#define DDIM     3072
#define ROW_TILE 16
#define NWAVES   8
#define NWG      (B_PAD / ROW_TILE)   /* 63 */
#define STG_DEPTH 4                   /* async double^2-buffer ring per wave */
#define CHUNK_ELEMS (16 * 32)         /* one 16x32 bf16 weight chunk = 1KB */

// ---------------------------------------------------------------------------
// WMMA helper (CDNA5 wave32, 16x16x32 bf16 -> f32)
// ---------------------------------------------------------------------------
__device__ __forceinline__ v8f wmma_bf16(v16bf a, v16bf b, v8f c) {
  return __builtin_amdgcn_wmma_f32_16x16x32_bf16(false, a, false, b, (short)0, c,
                                                 false, false);
}

// Load a 16x32 bf16 fragment (A layout; same math serves B stored [n][k]).
// Per the ISA 16-bit layout: lane&15 = row, lane>>4 selects k-half; two
// contiguous 16-byte loads per lane -> b128 loads (ds or global).
__device__ __forceinline__ v16bf load_frag16(const bf16* p0, int ldk) {
  const int lane = threadIdx.x & 31;
  const bf16* p = p0 + (lane & 15) * ldk + ((lane >> 4) << 3);
  union { v16bf w; v8bf h[2]; } u;
  u.h[0] = *(const v8bf*)(p);
  u.h[1] = *(const v8bf*)(p + 16);
  return u.w;
}

// ---------------------------------------------------------------------------
// Async weight streaming: copy one 16x32 bf16 chunk (16 rows x 64B) from
// global (row stride ldkG elements) into an LDS staging buffer ([row][k],
// ldk=32). 32 lanes x 16B x 2 instructions = 1KB. INST_OFFSET is added to
// BOTH the global and LDS addresses per the ISA, so one base serves both
// halves. Tracked with ASYNCcnt (2 per chunk), completes in order.
// ---------------------------------------------------------------------------
__device__ __forceinline__ void async_copy_chunk(const bf16* __restrict__ g,
                                                 int ldkG, bf16* lds_dst) {
  const int lane = threadIdx.x & 31;
  const int row  = lane >> 1;       // 0..15
  const int seg  = lane & 1;        // two 16B segments per row
  const bf16* gp = g + row * ldkG + seg * 8;
  const unsigned lo = (unsigned)(size_t)(lds_dst + row * 32 + seg * 8);
  asm volatile("global_load_async_to_lds_b128 %0, %1, off"
               :: "v"(lo), "v"(gp) : "memory");
  asm volatile("global_load_async_to_lds_b128 %0, %1, off offset:32"
               :: "v"(lo), "v"(gp) : "memory");
}

#define ASYNC_WAIT(n) asm volatile("s_wait_asynccnt " #n ::: "memory")

// In-order completion => chunk kk is done only when at most the chunks issued
// AFTER kk remain outstanding (2 async ops per chunk, issue-ahead of 3).
#define ASYNC_RING_WAIT(kk, steps)                                            \
  do {                                                                        \
    if      ((kk) + 3 < (steps)) ASYNC_WAIT(6);                               \
    else if ((kk) + 2 < (steps)) ASYNC_WAIT(4);                               \
    else if ((kk) + 1 < (steps)) ASYNC_WAIT(2);                               \
    else                         ASYNC_WAIT(0);                               \
  } while (0)

// ---------------------------------------------------------------------------
// Prep kernels: build z/dc (bf16, zero padded) and bf16 transposed weights
// ---------------------------------------------------------------------------
__global__ void prep_zdc(const float* __restrict__ c0, const float* __restrict__ c1,
                         const float* __restrict__ interior,
                         bf16* __restrict__ Z, bf16* __restrict__ DC,
                         float* __restrict__ out) {
  const int idx = blockIdx.x * blockDim.x + threadIdx.x;
  if (idx == 0) out[0] = 0.0f;
  if (idx >= B_PAD * MDIM) return;
  const int b = idx >> 5, m = idx & (MDIM - 1);
  if (b < B_ROWS) {
    const float cur = (b == 0) ? c0[m] : interior[(b - 1) * MDIM + m];
    const float nxt = (b < B_ROWS - 1) ? interior[b * MDIM + m] : c1[m];
    Z[idx]  = (bf16)cur;
    DC[idx] = (bf16)(nxt - cur);
  } else {
    Z[idx]  = (bf16)0.0f;
    DC[idx] = (bf16)0.0f;
  }
}

// Wt[n][k] = (bf16) W[k][n]   (W is KxN row-major fp32)
__global__ void prep_wt(const float* __restrict__ W, bf16* __restrict__ Wt,
                        int K, int N) {
  const int idx = blockIdx.x * blockDim.x + threadIdx.x;
  if (idx >= K * N) return;
  const int n = idx / K, k = idx - n * K;
  Wt[idx] = (bf16)W[(long)k * N + n];
}

// ---------------------------------------------------------------------------
// Fused JVP-energy kernel: one 16-row batch tile per workgroup, 8 wave32s.
// energy = sum_b || J_b dc_b ||^2 computed as a tangent (JVP) pass.
// ---------------------------------------------------------------------------
__global__ __launch_bounds__(256) void geodesic_energy_kernel(
    const bf16* __restrict__ Z, const bf16* __restrict__ DC,
    const bf16* __restrict__ W1t, const bf16* __restrict__ W2t,
    const bf16* __restrict__ W3t,
    const float* __restrict__ b1, const float* __restrict__ b2,
    float* __restrict__ out)
{
  extern __shared__ __align__(16) unsigned char smem_raw[];
  bf16* sZ   = (bf16*)smem_raw;                 // 16 x 32
  bf16* sDC  = sZ   + ROW_TILE * MDIM;          // 16 x 32
  bf16* sA1  = sDC  + ROW_TILE * MDIM;          // 16 x 512
  bf16* sDA1 = sA1  + ROW_TILE * H1DIM;         // 16 x 512
  bf16* sDA2 = sDA1 + ROW_TILE * H1DIM;         // 16 x 2048
  bf16* sStg = sDA2 + ROW_TILE * H2DIM;         // 8 waves x 4 x (16x32)

  const int tid  = threadIdx.x;
  const int lane = tid & 31;
  const int wave = tid >> 5;
  const int r0   = blockIdx.x * ROW_TILE;

  bf16* sW = sStg + wave * (STG_DEPTH * CHUNK_ELEMS);   // this wave's ring

  for (int i = tid; i < ROW_TILE * MDIM; i += 256) {
    sZ[i]  = Z [r0 * MDIM + i];
    sDC[i] = DC[r0 * MDIM + i];
  }
  __syncthreads();

  const int crow = (lane >> 4) << 3;   // C/D frag rows crow..crow+7
  const int ccol = lane & 15;          // C/D frag column

  // ---- Layer 1: h1 = z@W1 (+b1), dh1 = dc@W1 ; relu mask -> sA1 / sDA1
  {
    v16bf aZ  = load_frag16(sZ,  MDIM);
    v16bf aDC = load_frag16(sDC, MDIM);
    for (int jt = 0; jt < (H1DIM / 16) / NWAVES; ++jt) {
      const int jn = wave + jt * NWAVES;
      v16bf bW = load_frag16(W1t + jn * 16 * MDIM, MDIM);
      v8f h  = {};
      v8f dh = {};
      h  = wmma_bf16(aZ,  bW, h);
      dh = wmma_bf16(aDC, bW, dh);
      const float bias = b1[jn * 16 + ccol];
#pragma unroll
      for (int v = 0; v < 8; ++v) {
        const float hv = h[v] + bias;
        const bool  m  = hv > 0.0f;
        const int row  = crow + v;
        sA1 [row * H1DIM + jn * 16 + ccol] = (bf16)(m ? hv    : 0.0f);
        sDA1[row * H1DIM + jn * 16 + ccol] = (bf16)(m ? dh[v] : 0.0f);
      }
    }
  }
  __syncthreads();

  // ---- Layer 2: h2 = a1@W2 (+b2), dh2 = da1@W2 ; mask -> sDA2
  // W2 chunks streamed via async-to-LDS ring (ASYNCcnt), overlapped with WMMA.
  {
    const int KSTEPS = H1DIM / 32;                       // 16
    for (int jt = 0; jt < (H2DIM / 16) / NWAVES; ++jt) { // 16 N-tiles / wave
      const int jn = wave + jt * NWAVES;
      const bf16* wb = W2t + (long)jn * 16 * H1DIM;
      async_copy_chunk(wb + 0 * 32, H1DIM, sW + 0 * CHUNK_ELEMS);
      async_copy_chunk(wb + 1 * 32, H1DIM, sW + 1 * CHUNK_ELEMS);
      async_copy_chunk(wb + 2 * 32, H1DIM, sW + 2 * CHUNK_ELEMS);
      v8f h  = {};
      v8f dh = {};
#pragma unroll 4
      for (int kk = 0; kk < KSTEPS; ++kk) {
        if (kk + 3 < KSTEPS)
          async_copy_chunk(wb + (kk + 3) * 32, H1DIM,
                           sW + ((kk + 3) & (STG_DEPTH - 1)) * CHUNK_ELEMS);
        ASYNC_RING_WAIT(kk, KSTEPS);
        v16bf bW  = load_frag16(sW + (kk & (STG_DEPTH - 1)) * CHUNK_ELEMS, 32);
        v16bf aA  = load_frag16(sA1  + kk * 32, H1DIM);
        v16bf aDA = load_frag16(sDA1 + kk * 32, H1DIM);
        h  = wmma_bf16(aA,  bW, h);
        dh = wmma_bf16(aDA, bW, dh);
      }
      const float bias = b2[jn * 16 + ccol];
#pragma unroll
      for (int v = 0; v < 8; ++v) {
        const bool m = (h[v] + bias) > 0.0f;
        sDA2[(crow + v) * H2DIM + jn * 16 + ccol] = (bf16)(m ? dh[v] : 0.0f);
      }
    }
  }
  __syncthreads();

  // ---- Layer 3: dout = da2@W3 ; energy += sum(dout^2). Never materialized.
  float sumsq = 0.0f;
  {
    const int KSTEPS = H2DIM / 32;                       // 64
    for (int jt = 0; jt < (DDIM / 16) / NWAVES; ++jt) {  // 24 N-tiles / wave
      const int jn = wave + jt * NWAVES;
      const bf16* wb = W3t + (long)jn * 16 * H2DIM;
      __builtin_prefetch(wb, 0, 1);                      // global_prefetch_b8
      async_copy_chunk(wb + 0 * 32, H2DIM, sW + 0 * CHUNK_ELEMS);
      async_copy_chunk(wb + 1 * 32, H2DIM, sW + 1 * CHUNK_ELEMS);
      async_copy_chunk(wb + 2 * 32, H2DIM, sW + 2 * CHUNK_ELEMS);
      v8f acc = {};
#pragma unroll 4
      for (int kk = 0; kk < KSTEPS; ++kk) {
        if (kk + 3 < KSTEPS)
          async_copy_chunk(wb + (kk + 3) * 32, H2DIM,
                           sW + ((kk + 3) & (STG_DEPTH - 1)) * CHUNK_ELEMS);
        ASYNC_RING_WAIT(kk, KSTEPS);
        v16bf bW   = load_frag16(sW + (kk & (STG_DEPTH - 1)) * CHUNK_ELEMS, 32);
        v16bf aDA2 = load_frag16(sDA2 + kk * 32, H2DIM);
        acc = wmma_bf16(aDA2, bW, acc);
      }
#pragma unroll
      for (int v = 0; v < 8; ++v) sumsq += acc[v] * acc[v];
    }
  }

  // Block reduction + single atomic per workgroup.
  __shared__ float red[256];
  red[tid] = sumsq;
  __syncthreads();
  for (int s = 128; s > 0; s >>= 1) {
    if (tid < s) red[tid] += red[tid + s];
    __syncthreads();
  }
  if (tid == 0) atomicAdd(out, red[0]);
}

// ---------------------------------------------------------------------------
// Launch
// ---------------------------------------------------------------------------
extern "C" void kernel_launch(void* const* d_in, const int* in_sizes, int n_in,
                              void* d_out, int out_size, void* d_ws, size_t ws_size,
                              hipStream_t stream) {
  (void)in_sizes; (void)n_in; (void)out_size; (void)ws_size;
  const float* c0       = (const float*)d_in[0];
  const float* c1       = (const float*)d_in[1];
  const float* interior = (const float*)d_in[2];
  const float* W1       = (const float*)d_in[3];
  const float* b1       = (const float*)d_in[4];
  const float* W2       = (const float*)d_in[5];
  const float* b2       = (const float*)d_in[6];
  const float* W3       = (const float*)d_in[7];
  // d_in[8] (b3) does not affect the Jacobian energy.

  unsigned char* ws = (unsigned char*)d_ws;
  bf16* Z   = (bf16*)(ws + 0);         // 1008*32 bf16
  bf16* DC  = (bf16*)(ws + 65536);     // 1008*32 bf16
  bf16* W1t = (bf16*)(ws + 131072);    // 512*32
  bf16* W2t = (bf16*)(ws + 163840);    // 2048*512
  bf16* W3t = (bf16*)(ws + 2260992);   // 3072*2048  (ends ~14.2 MB)
  float* out = (float*)d_out;

  prep_zdc<<<(B_PAD * MDIM + 255) / 256, 256, 0, stream>>>(c0, c1, interior, Z, DC, out);
  prep_wt <<<(MDIM  * H1DIM + 255) / 256, 256, 0, stream>>>(W1, W1t, MDIM,  H1DIM);
  prep_wt <<<(H1DIM * H2DIM + 255) / 256, 256, 0, stream>>>(W2, W2t, H1DIM, H2DIM);
  prep_wt <<<(H2DIM * DDIM  + 255) / 256, 256, 0, stream>>>(W3, W3t, H2DIM, DDIM);

  const size_t smem = (size_t)(2 * ROW_TILE * MDIM +          // sZ + sDC
                               2 * ROW_TILE * H1DIM +         // sA1 + sDA1
                               ROW_TILE * H2DIM +             // sDA2
                               NWAVES * STG_DEPTH * CHUNK_ELEMS) // staging ring
                      * sizeof(bf16);                         // 133,120 B
  geodesic_energy_kernel<<<NWG, 256, smem, stream>>>(Z, DC, W1t, W2t, W3t, b1, b2, out);
}
